// GuidedSegmentTransformer_3934190043521
// MI455X (gfx1250) — compile-verified, matrix operand order checked by reading, not run on hardware
//
#include <hip/hip_runtime.h>
#include <hip/hip_bf16.h>

// MI455X / gfx1250, wave32. Heavy math via v_wmma_f32_16x16x32_f16,
// pure-copy LDS staging via global_load_async_to_lds_* (ASYNCcnt).
typedef __attribute__((ext_vector_type(16))) _Float16 v16h;
typedef __attribute__((ext_vector_type(8)))  float    v8f;

__device__ __forceinline__ v8f wmma_f16(v16h a, v16h b, v8f c) {
  // 8 args: (neg_a, A, neg_b, B, c_mod, C, reuse_a, reuse_b)
  return __builtin_amdgcn_wmma_f32_16x16x32_f16(false, a, false, b, (short)0, c, false, false);
}

// Async global->LDS copies (CDNA5, tracked by ASYNCcnt).
// lds_off = byte offset within the workgroup's LDS allocation (low 32 bits of
// the generic shared pointer); VDST operand carries the LDS address.
__device__ __forceinline__ void async_copy_b128(unsigned lds_off, const void* g) {
  asm volatile("global_load_async_to_lds_b128 %0, %1, off"
               :: "v"(lds_off), "v"((unsigned long long)g) : "memory");
}
__device__ __forceinline__ void async_copy_b32(unsigned lds_off, const void* g) {
  asm volatile("global_load_async_to_lds_b32 %0, %1, off"
               :: "v"(lds_off), "v"((unsigned long long)g) : "memory");
}
__device__ __forceinline__ void wait_async0() {
  asm volatile("s_wait_asynccnt 0x0" ::: "memory");
}
__device__ __forceinline__ unsigned lds_off32(const void* p) {
  return (unsigned)(size_t)p;   // LDS aperture: low 32 bits = LDS byte offset
}

// Load a 16x32 f16 A-fragment (or B-fragment from an [N][K] transposed tile).
// CDNA5 16-bit A layout: lanes 0-15 -> rows M=0..15 with K groups {0-7,16-23},
// lanes 16-31 -> same rows with K groups {8-15,24-31}; each VGPR holds K,K+1 packed.
__device__ __forceinline__ v16h load_frag16(const _Float16* base, int pitch) {
  const int lane = threadIdx.x & 31;
  const int row  = lane & 15;
  const int hi   = (lane >> 4) & 1;
  union { unsigned int u[8]; v16h h; } f;
  const unsigned int* p = (const unsigned int*)(base + row * pitch);
#pragma unroll
  for (int v = 0; v < 8; ++v) {
    const int kb = ((v & 4) << 2) | (hi << 3) | ((v & 3) << 1); // K base (even)
    f.u[v] = p[kb >> 1];
  }
  return f.h;
}

__device__ __forceinline__ float block_sum256(float v, float* sm) {
  const int t = threadIdx.x;
  sm[t] = v; __syncthreads();
  for (int s = 128; s > 0; s >>= 1) { if (t < s) sm[t] += sm[t + s]; __syncthreads(); }
  float r = sm[0]; __syncthreads();
  return r;
}
__device__ __forceinline__ float block_max256(float v, float* sm) {
  const int t = threadIdx.x;
  sm[t] = v; __syncthreads();
  for (int s = 128; s > 0; s >>= 1) { if (t < s) sm[t] = fmaxf(sm[t], sm[t + s]); __syncthreads(); }
  float r = sm[0]; __syncthreads();
  return r;
}

// ---------------------------------------------------------------------------
// Generic WMMA GEMM: C[M=1024,N] = A[M,K] * B[K,N] + bias (+ pos/residual)
// Grid: (N/64, M/128), 256 threads (8 waves: 4 in M x 2 in N).
// addMode: 0 none, 1 += add[(row mod 512)*ldc+col] (pos), 2 += add[row*ldc+col]
// ---------------------------------------------------------------------------
__global__ __launch_bounds__(256)
void k_gemm(const float* __restrict__ A, int lda,
            const float* __restrict__ Bm, int ldb,
            const float* __restrict__ bias,
            const float* __restrict__ add, int addMode,
            float* __restrict__ C, int ldc, int K) {
  __shared__ __align__(16) _Float16 As[128][32];
  __shared__ __align__(16) _Float16 Bs[64][32];   // transposed [n][k]
  const int tid = threadIdx.x, lane = tid & 31, wid = tid >> 5;
  const int wm = wid & 3, wn = wid >> 2;
  const int m0 = blockIdx.y * 128, n0 = blockIdx.x * 64;
  v8f acc[2][2] = {};
  const int nk = K >> 5;
  for (int kt = 0; kt < nk; ++kt) {
    for (int i = tid; i < 128 * 32; i += 256) {
      const int r = i >> 5, c = i & 31;
      As[r][c] = (_Float16)A[(size_t)(m0 + r) * lda + kt * 32 + c];
    }
    for (int i = tid; i < 64 * 32; i += 256) {
      const int k = i >> 6, n = i & 63;   // coalesced global, transposed LDS store
      Bs[n][k] = (_Float16)Bm[(size_t)(kt * 32 + k) * ldb + n0 + n];
    }
    __syncthreads();
    if (kt + 1 < nk)  // -> global_prefetch_b8
      __builtin_prefetch(&A[(size_t)(m0 + (tid >> 1)) * lda + (kt + 1) * 32], 0, 0);
    v16h af[2], bf[2];
#pragma unroll
    for (int mi = 0; mi < 2; ++mi) af[mi] = load_frag16(&As[wm * 32 + mi * 16][0], 32);
#pragma unroll
    for (int ni = 0; ni < 2; ++ni) bf[ni] = load_frag16(&Bs[wn * 32 + ni * 16][0], 32);
#pragma unroll
    for (int mi = 0; mi < 2; ++mi)
#pragma unroll
      for (int ni = 0; ni < 2; ++ni)
        acc[mi][ni] = wmma_f16(af[mi], bf[ni], acc[mi][ni]);
    __syncthreads();
  }
  const int hi = (lane >> 4) & 1, nc = lane & 15;
#pragma unroll
  for (int mi = 0; mi < 2; ++mi)
#pragma unroll
    for (int ni = 0; ni < 2; ++ni)
#pragma unroll
      for (int r = 0; r < 8; ++r) {
        const int row = m0 + wm * 32 + mi * 16 + r + hi * 8;
        const int col = n0 + wn * 32 + ni * 16 + nc;
        float v = acc[mi][ni][r];
        if (bias) v += bias[col];
        if (addMode == 1)      v += add[(size_t)(row & 511) * ldc + col];
        else if (addMode == 2) v += add[(size_t)row * ldc + col];
        C[(size_t)row * ldc + col] = v;
      }
}

// ---------------------------------------------------------------------------
// scores[b,h,q,k] = (Q . K) / sqrt(D), masked.  Per (b,h): 512x512, K=32.
// Grid: (512/128, 512/64, 16); 8 waves: 2 in M x 4 in N.
// ---------------------------------------------------------------------------
__global__ __launch_bounds__(256)
void k_scores(const float* __restrict__ Q, const float* __restrict__ Kb,
              const unsigned char* __restrict__ pm, float* __restrict__ sc) {
  __shared__ __align__(16) _Float16 Qs[64][32];
  __shared__ __align__(16) _Float16 Ks[128][32];
  const int tid = threadIdx.x, lane = tid & 31, wid = tid >> 5;
  const int wm = wid & 1, wn = wid >> 1;
  const int bh = blockIdx.z, b = bh >> 3, h = bh & 7;
  const int m0 = blockIdx.y * 64, n0 = blockIdx.x * 128;
  for (int i = tid; i < 64 * 32; i += 256) {
    const int r = i >> 5, k = i & 31;
    Qs[r][k] = (_Float16)Q[(size_t)(b * 512 + m0 + r) * 256 + h * 32 + k];
  }
  for (int i = tid; i < 128 * 32; i += 256) {
    const int r = i >> 5, k = i & 31;
    Ks[r][k] = (_Float16)Kb[(size_t)(b * 512 + n0 + r) * 256 + h * 32 + k];
  }
  __syncthreads();
  v16h af[2], bf[2];
#pragma unroll
  for (int mi = 0; mi < 2; ++mi) af[mi] = load_frag16(&Qs[wm * 32 + mi * 16][0], 32);
#pragma unroll
  for (int ni = 0; ni < 2; ++ni) bf[ni] = load_frag16(&Ks[wn * 32 + ni * 16][0], 32);
  v8f acc[2][2] = {};
#pragma unroll
  for (int mi = 0; mi < 2; ++mi)
#pragma unroll
    for (int ni = 0; ni < 2; ++ni)
      acc[mi][ni] = wmma_f16(af[mi], bf[ni], acc[mi][ni]);
  const int hi = (lane >> 4) & 1, nc = lane & 15;
#pragma unroll
  for (int mi = 0; mi < 2; ++mi)
#pragma unroll
    for (int ni = 0; ni < 2; ++ni)
#pragma unroll
      for (int r = 0; r < 8; ++r) {
        const int qr = m0 + wm * 32 + mi * 16 + r + hi * 8;
        const int kc = n0 + wn * 32 + ni * 16 + nc;
        float v = acc[mi][ni][r] * 0.0625f;          // 1/sqrt(256)
        if (pm[b * 512 + kc]) v = -1e30f;
        sc[((size_t)bh * 512 + qr) * 512 + kc] = v;
      }
}

// Softmax over 512 keys; emits probabilities directly as f16 (probs are in
// [0,1] -> f16 is plenty), so attn@V can stage them with pure async copies.
__global__ __launch_bounds__(256)
void k_softmax(const float* __restrict__ sc, _Float16* __restrict__ ph) {
  __shared__ float red[256];
  const int t = threadIdx.x;
  const float* row = sc + (size_t)blockIdx.x * 512;
  _Float16*   orow = ph + (size_t)blockIdx.x * 512;
  const float v0 = row[t], v1 = row[t + 256];
  const float m = block_max256(fmaxf(v0, v1), red);
  const float e0 = __expf(v0 - m), e1 = __expf(v1 - m);
  const float inv = 1.0f / block_sum256(e0 + e1, red);
  orow[t] = (_Float16)(e0 * inv); orow[t + 256] = (_Float16)(e1 * inv);
}

// att[b,s,h*32+d] = attn @ V.  Per (b,h): M=512, N=32, K=512.
// Grid: (512/128, 16); 8 waves, each owns 16 rows.
// Probability tiles are f16 in global -> staged with async b128 copies.
__global__ __launch_bounds__(256)
void k_attnv(const _Float16* __restrict__ ph, const float* __restrict__ V,
             float* __restrict__ att) {
  __shared__ __align__(16) _Float16 Ps[128][32];
  __shared__ __align__(16) _Float16 Vs[32][32];   // transposed [n][k]
  const int tid = threadIdx.x, lane = tid & 31, wid = tid >> 5;
  const int bh = blockIdx.y, b = bh >> 3, h = bh & 7;
  const int m0 = blockIdx.x * 128;
  v8f acc[2] = {};
  for (int kt = 0; kt < 16; ++kt) {
    // Ps[r][0..31] <- ph row slab: contiguous 64B per row, 16B per async op.
    for (int i = tid; i < 512; i += 256) {
      const int r = i >> 2, seg = i & 3;
      async_copy_b128(lds_off32(&Ps[r][seg * 8]),
                      ph + ((size_t)bh * 512 + m0 + r) * 512 + kt * 32 + seg * 8);
    }
    // Vs needs fp32->f16 conversion: regular staging, overlapped with async.
    for (int i = tid; i < 32 * 32; i += 256) {
      const int n = i >> 5, k = i & 31;
      Vs[n][k] = (_Float16)V[(size_t)(b * 512 + kt * 32 + k) * 256 + h * 32 + n];
    }
    wait_async0();          // this wave's ASYNCcnt drained ...
    __syncthreads();        // ... then cross-wave visibility via barrier
    const v16h a = load_frag16(&Ps[wid * 16][0], 32);
#pragma unroll
    for (int ni = 0; ni < 2; ++ni) {
      const v16h bfr = load_frag16(&Vs[ni * 16][0], 32);
      acc[ni] = wmma_f16(a, bfr, acc[ni]);
    }
    __syncthreads();
  }
  const int hi = (lane >> 4) & 1, nc = lane & 15;
#pragma unroll
  for (int ni = 0; ni < 2; ++ni)
#pragma unroll
    for (int r = 0; r < 8; ++r) {
      const int row = m0 + wid * 16 + r + hi * 8;
      att[(size_t)(b * 512 + row) * 256 + h * 32 + ni * 16 + nc] = acc[ni][r];
    }
}

__global__ __launch_bounds__(256)
void k_ln(const float* __restrict__ in, const float* __restrict__ g,
          const float* __restrict__ be, float* __restrict__ out) {
  __shared__ float red[256];
  const int t = threadIdx.x; const size_t row = blockIdx.x;
  const float v = in[row * 256 + t];
  const float m = block_sum256(v, red) * (1.0f / 256.0f);
  const float d = v - m;
  const float var = block_sum256(d * d, red) * (1.0f / 256.0f);
  out[row * 256 + t] = d * rsqrtf(var + 1e-5f) * g[t] + be[t];
}

// pw[b,i,j] = exp(-(|xi|^2+|xj|^2-2 xi.xj)/256), masked to 0.
__global__ __launch_bounds__(256)
void k_pairwise(const float* __restrict__ x, const unsigned char* __restrict__ pm,
                float* __restrict__ pw) {
  __shared__ float xi[256];
  const int t = threadIdx.x, bi = blockIdx.x;       // b*512 + i
  const int b = bi >> 9, i = bi & 511;
  xi[t] = x[(size_t)bi * 256 + t];
  __syncthreads();
  float sni = 0.f;
  for (int k = 0; k < 256; ++k) sni += xi[k] * xi[k];
  const bool mski = pm[b * 512 + i] != 0;
  for (int jj = 0; jj < 2; ++jj) {
    const int j = t + jj * 256;
    const float* xj = x + (size_t)(b * 512 + j) * 256;
    float dot = 0.f, snj = 0.f;
    for (int k = 0; k < 256; ++k) { const float a = xj[k]; dot += xi[k] * a; snj += a * a; }
    float v = __expf(-(sni + snj - 2.f * dot) * (1.0f / 256.0f));
    if (mski || pm[b * 512 + j]) v = 0.f;
    pw[(size_t)bi * 512 + j] = v;
  }
}

// c2_w (256,128,3) -> f16 [tap][co][ci] for direct A-fragment loads.
__global__ void k_w2prep(const float* __restrict__ c2w, _Float16* __restrict__ w2h) {
  const int idx = blockIdx.x * 256 + threadIdx.x;
  if (idx >= 3 * 256 * 128) return;
  const int t = idx / 32768, rem = idx % 32768;
  const int co = rem >> 7, ci = rem & 127;
  w2h[idx] = (_Float16)c2w[(co * 128 + ci) * 3 + t];
}

// Structure branch, one WG per (b,s): conv1 -> LDS y1T -> conv2 as 3 shifted
// WMMA GEMMs (256x64 output slabs) -> fused bias/ReLU/column-mean -> +pos.
// The fp32 pw row is a pure copy -> async b32 to LDS.
__global__ __launch_bounds__(256)
void k_conv(const float* __restrict__ pw, const float* __restrict__ c1w,
            const float* __restrict__ c1b, const _Float16* __restrict__ w2h,
            const float* __restrict__ c2b, const float* __restrict__ pos,
            float* __restrict__ x2) {
  __shared__ float    pwl[514];          // padded conv1 input row
  __shared__ __align__(16) _Float16 y1T[66][128]; // [l - l0 + 1][ci], with halo
  __shared__ float    meanl[256];
  __shared__ float    w1l[384];
  __shared__ float    b1l[128];
  __shared__ float    b2l[256];
  const int tid = threadIdx.x, lane = tid & 31, w = tid >> 5;
  const int bs = blockIdx.x, s = bs & 511;
  for (int i = tid; i < 512; i += 256)   // async: fp32 row copy, 4B aligned halo
    async_copy_b32(lds_off32(&pwl[1 + i]), pw + (size_t)bs * 512 + i);
  if (tid == 0) { pwl[0] = 0.f; pwl[513] = 0.f; }
  for (int i = tid; i < 384; i += 256) w1l[i] = c1w[i];
  if (tid < 128) b1l[tid] = c1b[tid];
  b2l[tid] = c2b[tid];
  meanl[tid] = 0.f;
  wait_async0();
  const int hi = (lane >> 4) & 1;
  float macc[2][8];
#pragma unroll
  for (int mi = 0; mi < 2; ++mi)
#pragma unroll
    for (int r = 0; r < 8; ++r) macc[mi][r] = 0.f;

  for (int lt = 0; lt < 8; ++lt) {
    const int l0 = lt * 64;
    __syncthreads();
    for (int i = tid; i < 66 * 128; i += 256) {       // conv1 + ReLU into y1T
      const int j = i >> 7, ci = i & 127;
      const int l = l0 + j - 1;
      float v = 0.f;
      if (l >= 0 && l < 512)
        v = fmaxf(b1l[ci] + w1l[ci * 3 + 0] * pwl[l] + w1l[ci * 3 + 1] * pwl[l + 1]
                          + w1l[ci * 3 + 2] * pwl[l + 2], 0.f);
      y1T[j][ci] = (_Float16)v;
    }
    __syncthreads();
    v8f acc[2][4] = {};
    for (int t = 0; t < 3; ++t) {                     // tap
      for (int kk = 0; kk < 4; ++kk) {                // K chunks of 32 (ci)
        const v16h a0 = load_frag16(w2h + t * 32768 + (w * 32 + 0)  * 128 + kk * 32, 128);
        const v16h a1 = load_frag16(w2h + t * 32768 + (w * 32 + 16) * 128 + kk * 32, 128);
#pragma unroll
        for (int ni = 0; ni < 4; ++ni) {
          const v16h bfr = load_frag16(&y1T[ni * 16 + t][kk * 32], 128);
          acc[0][ni] = wmma_f16(a0, bfr, acc[0][ni]);
          acc[1][ni] = wmma_f16(a1, bfr, acc[1][ni]);
        }
      }
    }
#pragma unroll
    for (int mi = 0; mi < 2; ++mi)
#pragma unroll
      for (int ni = 0; ni < 4; ++ni)
#pragma unroll
        for (int r = 0; r < 8; ++r) {
          const int co = w * 32 + mi * 16 + r + hi * 8;
          macc[mi][r] += fmaxf(acc[mi][ni][r] + b2l[co], 0.f);
        }
  }
#pragma unroll
  for (int mi = 0; mi < 2; ++mi)
#pragma unroll
    for (int r = 0; r < 8; ++r)
      atomicAdd(&meanl[w * 32 + mi * 16 + r + hi * 8], macc[mi][r]);  // ds_add_f32
  __syncthreads();
  x2[(size_t)bs * 256 + tid] = meanl[tid] * (1.0f / 512.0f) + pos[s * 256 + tid];
}

// Masked-mean pooling + 3-layer MLP head with layernorms. B=2 -> 2 blocks.
__global__ __launch_bounds__(256)
void k_pool_head(const float* __restrict__ x1, const float* __restrict__ x2,
                 const unsigned char* __restrict__ pm,
                 const float* __restrict__ h1w, const float* __restrict__ h1b,
                 const float* __restrict__ g1, const float* __restrict__ be1,
                 const float* __restrict__ h2w, const float* __restrict__ h2b,
                 const float* __restrict__ g2, const float* __restrict__ be2,
                 const float* __restrict__ h3w, const float* __restrict__ h3b,
                 float* __restrict__ out) {
  __shared__ float hcat[512];
  __shared__ float a1[256];
  __shared__ float zb[256];
  const int t = threadIdx.x, b = blockIdx.x;
  float cnt = 0.f, s1 = 0.f, s2 = 0.f;
  for (int s = 0; s < 512; ++s) {
    const float m = pm[b * 512 + s] ? 0.f : 1.f;
    cnt += m;
    s1 += m * x1[(size_t)(b * 512 + s) * 256 + t];
    s2 += m * x2[(size_t)(b * 512 + s) * 256 + t];
  }
  hcat[t] = s1 / cnt; hcat[256 + t] = s2 / cnt;
  __syncthreads();
  float z = h1b[t];
  for (int j = 0; j < 512; ++j) z += hcat[j] * h1w[j * 256 + t];
  zb[t] = z; __syncthreads();
  float m1 = 0.f; for (int j = 0; j < 256; ++j) m1 += zb[j];
  m1 *= (1.0f / 256.0f);
  float v1 = 0.f; for (int j = 0; j < 256; ++j) { const float d = zb[j] - m1; v1 += d * d; }
  v1 *= (1.0f / 256.0f);
  a1[t] = fmaxf((z - m1) * rsqrtf(v1 + 1e-5f) * g1[t] + be1[t], 0.f);
  __syncthreads();
  float z2 = 0.f;
  if (t < 128) {
    z2 = h2b[t];
    for (int j = 0; j < 256; ++j) z2 += a1[j] * h2w[j * 128 + t];
    zb[t] = z2;
  }
  __syncthreads();
  float m2 = 0.f; for (int j = 0; j < 128; ++j) m2 += zb[j];
  m2 *= (1.0f / 128.0f);
  float v2 = 0.f; for (int j = 0; j < 128; ++j) { const float d = zb[j] - m2; v2 += d * d; }
  v2 *= (1.0f / 128.0f);
  __syncthreads();
  if (t < 128) zb[t] = fmaxf((z2 - m2) * rsqrtf(v2 + 1e-5f) * g2[t] + be2[t], 0.f);
  __syncthreads();
  if (t == 0) {
    float o = h3b[0];
    for (int j = 0; j < 128; ++j) o += zb[j] * h3w[j];
    out[b] = o;
  }
}

extern "C" void kernel_launch(void* const* d_in, const int* in_sizes, int n_in,
                              void* d_out, int out_size, void* d_ws, size_t ws_size,
                              hipStream_t stream) {
  (void)in_sizes; (void)n_in; (void)out_size; (void)ws_size;
  const float* x   = (const float*)d_in[0];
  const unsigned char* pm = (const unsigned char*)d_in[1];
  const float* Win = (const float*)d_in[2];
  const float* bin = (const float*)d_in[3];
  const float* pos = (const float*)d_in[4];
  const float* Wq  = (const float*)d_in[5];
  const float* Wk  = (const float*)d_in[6];
  const float* Wv  = (const float*)d_in[7];
  const float* Wo  = (const float*)d_in[8];
  const float* bq  = (const float*)d_in[9];
  const float* bk  = (const float*)d_in[10];
  const float* bv  = (const float*)d_in[11];
  const float* bo  = (const float*)d_in[12];
  const float* lng = (const float*)d_in[13];
  const float* lnb = (const float*)d_in[14];
  const float* c1w = (const float*)d_in[15];
  const float* c1b = (const float*)d_in[16];
  const float* c2w = (const float*)d_in[17];
  const float* c2b = (const float*)d_in[18];
  const float* h1w = (const float*)d_in[19];
  const float* h1b = (const float*)d_in[20];
  const float* g1  = (const float*)d_in[21];
  const float* be1 = (const float*)d_in[22];
  const float* h2w = (const float*)d_in[23];
  const float* h2b = (const float*)d_in[24];
  const float* g2  = (const float*)d_in[25];
  const float* be2 = (const float*)d_in[26];
  const float* h3w = (const float*)d_in[27];
  const float* h3b = (const float*)d_in[28];

  float* ws  = (float*)d_ws;
  float* x1  = ws;                   // 1024*256
  float* tmp = x1  + 262144;
  float* Qb  = tmp + 262144;
  float* Kb  = Qb  + 262144;
  float* Vb  = Kb  + 262144;
  float* att = Vb  + 262144;
  float* pw  = att + 262144;         // 2*512*512
  float* x2  = pw  + 524288;
  float* sc  = x2  + 262144;         // 16*512*512
  _Float16* w2h = (_Float16*)(sc + 4194304);       // 3*256*128 f16
  _Float16* ph  = w2h + 3 * 256 * 128;             // 16*512*512 f16 probs

  const dim3 blk(256);
  k_w2prep  <<<dim3(384),  blk, 0, stream>>>(c2w, w2h);
  k_pairwise<<<dim3(1024), blk, 0, stream>>>(x, pm, pw);
  // x1 = x @ W_in + b_in + pos
  k_gemm<<<dim3(4, 8), blk, 0, stream>>>(x, 256, Win, 256, bin, pos, 1, x1, 256, 256);
  for (int i = 0; i < 4; ++i) {
    k_gemm<<<dim3(4, 8), blk, 0, stream>>>(x1, 256, Wq + (size_t)i * 65536, 256,
                                           bq + i * 256, nullptr, 0, Qb, 256, 256);
    k_gemm<<<dim3(4, 8), blk, 0, stream>>>(x1, 256, Wk + (size_t)i * 65536, 256,
                                           bk + i * 256, nullptr, 0, Kb, 256, 256);
    k_gemm<<<dim3(4, 8), blk, 0, stream>>>(x1, 256, Wv + (size_t)i * 65536, 256,
                                           bv + i * 256, nullptr, 0, Vb, 256, 256);
    k_scores <<<dim3(4, 8, 16), blk, 0, stream>>>(Qb, Kb, pm, sc);
    k_softmax<<<dim3(8192),     blk, 0, stream>>>(sc, ph);
    k_attnv  <<<dim3(4, 16),    blk, 0, stream>>>(ph, Vb, att);
    k_gemm<<<dim3(4, 8), blk, 0, stream>>>(att, 256, Wo + (size_t)i * 65536, 256,
                                           bo + i * 256, x1, 2, tmp, 256, 256);
    k_ln<<<dim3(1024), blk, 0, stream>>>(tmp, lng + i * 256, lnb + i * 256, x1);
  }
  k_conv<<<dim3(1024), blk, 0, stream>>>(pw, c1w, c1b, w2h, c2b, pos, x2);
  k_pool_head<<<dim3(2), blk, 0, stream>>>(x1, x2, pm, h1w, h1b, g1, be1,
                                           h2w, h2b, g2, be2, h3w, h3b,
                                           (float*)d_out);
}
